// Attention_3264175145451
// MI455X (gfx1250) — matrix operand order repaired and checked by reading, not compiled, not run on hardware
//
#include <hip/hip_runtime.h>
#include <cstdint>

// ---------------------------------------------------------------------------
// Types matching CDNA5 WMMA operand layouts (wave32)
// ---------------------------------------------------------------------------
typedef __attribute__((ext_vector_type(16))) __bf16 v16bf;
typedef __attribute__((ext_vector_type(8)))  __bf16 v8bf;
typedef __attribute__((ext_vector_type(8)))  float  v8f;

#define DIM    512
#define HEADS  8
#define HD     64
#define NTOK   1024
#define BATCH  16
#define MTOT   (BATCH * NTOK)   // 16384 rows
#define KPAD   72               // LDS tile row stride in elements (144 B):
                                // 36-dword bank stepping -> conflict-free b128

union AF {                       // A-fragment: two contiguous 16B halves
    v16bf v;
    struct { v8bf lo, hi; } h;
};

static __device__ __forceinline__ v16bf ldv16(const __bf16* p) {
    return *reinterpret_cast<const v16bf*>(p);   // 32B aligned by construction
}
static __device__ __forceinline__ v8bf ldv8(const __bf16* p) {
    return *reinterpret_cast<const v8bf*>(p);    // 16B aligned by construction
}
// LDS reads: assemble 16 consecutive bf16 from two 16B-aligned halves
static __device__ __forceinline__ v16bf ldv16_lds(const __bf16* p) {
    AF t;
    t.h.lo = *reinterpret_cast<const v8bf*>(p);
    t.h.hi = *reinterpret_cast<const v8bf*>(p + 8);
    return t.v;
}
static __device__ __forceinline__ v8f wmma_bf16(v16bf a, v16bf b, v8f c) {
    return __builtin_amdgcn_wmma_f32_16x16x32_bf16(
        /*neg_a=*/false, a, /*neg_b=*/false, b,
        /*c_mod=*/(short)0, c, /*reuse_a=*/false, /*reuse_b=*/false);
}

// gfx1250 async copy global->LDS (ASYNCcnt-tracked). LDS byte address is the
// low 32 bits of the generic pointer to a __shared__ object.
static __device__ __forceinline__ uint32_t lds_off(const void* p) {
    return (uint32_t)(uintptr_t)p;
}
#define ASYNC_LD_B128(ldsaddr, gaddr)                                         \
    asm volatile("global_load_async_to_lds_b128 %0, %1, off"                  \
                 :: "v"(ldsaddr), "v"(gaddr) : "memory")

// ---------------------------------------------------------------------------
// Conversion kernels: fp32 -> bf16 (x straight, weights transposed to [N][K])
// ---------------------------------------------------------------------------
__global__ void cvt_x_kernel(const float* __restrict__ x, __bf16* __restrict__ xb, int n) {
    int i = blockIdx.x * blockDim.x + threadIdx.x;
    int stride = gridDim.x * blockDim.x;
    for (; i < n; i += stride) xb[i] = (__bf16)x[i];
}

// w: [K][N] fp32  ->  wt: [N][K] bf16
__global__ void cvt_wT_kernel(const float* __restrict__ w, __bf16* __restrict__ wt,
                              int K, int N) {
    int i = blockIdx.x * blockDim.x + threadIdx.x;
    int stride = gridDim.x * blockDim.x;
    int total = K * N;
    for (; i < total; i += stride) {
        int f = i / K;           // output row (N index)
        int k = i - f * K;       // output col (K index)
        wt[i] = (__bf16)w[(size_t)k * N + f];
    }
}

// ---------------------------------------------------------------------------
// QKV projection: [16384,512] x [512,1536] + bias, scatter to Q / K / V^T bf16
// One wave computes a 16(M) x 64(N) tile; each 64-wide N group is exactly one
// of {Q,K,V} of one head (192 = 3*64 channels per head).
// ---------------------------------------------------------------------------
__global__ __launch_bounds__(128)
void qkv_gemm_kernel(const __bf16* __restrict__ xb,
                     const __bf16* __restrict__ wqkvT,
                     const float*  __restrict__ bqkv,
                     __bf16* __restrict__ Q,
                     __bf16* __restrict__ K,
                     __bf16* __restrict__ Vt) {
    const int lane  = threadIdx.x & 31;
    const int wid   = threadIdx.x >> 5;
    const int half  = lane >> 4;
    const int l16   = lane & 15;
    const int wg    = blockIdx.x * 4 + wid;
    const int mtile = wg / 24;          // 0..1023
    const int n64   = wg - mtile * 24;  // 0..23

    const int arow = mtile * 16 + l16;

    v8f acc[4] = {v8f{}, v8f{}, v8f{}, v8f{}};

    for (int kk = 0; kk < DIM; kk += 32) {
        AF a;
        const __bf16* ap = xb + (size_t)arow * DIM + kk;
        a.h.lo = ldv8(ap + half * 16);
        a.h.hi = ldv8(ap + 8 + half * 16);
#pragma unroll
        for (int t = 0; t < 4; ++t) {
            const int fcol = n64 * 64 + t * 16 + l16;
            v16bf b = ldv16(wqkvT + (size_t)fcol * DIM + kk + half * 16);
            acc[t] = wmma_bf16(a.v, b, acc[t]);
        }
    }

    const int comp = n64 % 3;   // 0=Q 1=K 2=V
    const int head = n64 / 3;
#pragma unroll
    for (int t = 0; t < 4; ++t) {
        const int dd     = t * 16 + l16;
        const float bias = bqkv[head * 192 + comp * 64 + dd];
#pragma unroll
        for (int r = 0; r < 8; ++r) {
            const int row   = mtile * 16 + r + half * 8;
            const int token = row & (NTOK - 1);
            const int bidx  = row >> 10;
            const size_t bh = (size_t)(bidx * HEADS + head);
            const float v   = acc[t][r] + bias;
            if (comp == 0)      Q [(bh * NTOK + token) * HD + dd] = (__bf16)(v * 0.125f);
            else if (comp == 1) K [(bh * NTOK + token) * HD + dd] = (__bf16)v;
            else                Vt[(bh * HD + dd) * NTOK + token] = (__bf16)v;
        }
    }
}

// ---------------------------------------------------------------------------
// Flash attention, LDS-cooperative version.
// One block = 8 waves = 8 query tiles (128 queries) of one (batch, head).
// K/V streamed in 64-key chunks into double-buffered LDS tiles using
// global_load_async_to_lds_b128 (ASYNCcnt), overlapping copy of chunk i+1
// with compute of chunk i. All B-fragments come from LDS (ds_load_b128);
// P is transposed C->A layout through a per-wave LDS slab (s_wait_dscnt).
// ---------------------------------------------------------------------------
__global__ __launch_bounds__(256)
void attn_kernel(const __bf16* __restrict__ Q,
                 const __bf16* __restrict__ K,
                 const __bf16* __restrict__ Vt,
                 __bf16* __restrict__ aout) {
    __shared__ __bf16 Kbuf[2][64 * KPAD];   // [key][dk]   (2 x 9 KB)
    __shared__ __bf16 Vbuf[2][64 * KPAD];   // [dd][key]   (2 x 9 KB)
    __shared__ __bf16 Pbuf[8][16 * KPAD];   // per-wave transpose slab (18 KB)

    const int tid  = threadIdx.x;
    const int lane = tid & 31;
    const int wid  = tid >> 5;
    const int half = lane >> 4;
    const int l16  = lane & 15;

    const int bh    = blockIdx.x >> 3;              // 0..127
    const int qtile = (blockIdx.x & 7) * 8 + wid;   // 0..63

    const __bf16* Qb = Q  + (size_t)bh * NTOK * HD;
    const __bf16* Kb = K  + (size_t)bh * NTOK * HD;
    const __bf16* Vb = Vt + (size_t)bh * HD * NTOK;

    // async-copy work split: 256 threads cover 64 rows x 128 B (4 x 32 B segs)
    const int crow = tid >> 2;   // 0..63
    const int cseg = tid & 3;    // 0..3

    // Q A-fragments (registers for the whole kernel)
    AF qa0, qa1;
    {
        const __bf16* qp = Qb + (size_t)(qtile * 16 + l16) * HD;
        qa0.h.lo = ldv8(qp +      half * 16);
        qa0.h.hi = ldv8(qp +  8 + half * 16);
        qa1.h.lo = ldv8(qp + 32 + half * 16);
        qa1.h.hi = ldv8(qp + 40 + half * 16);
    }

    v8f O[4] = {v8f{}, v8f{}, v8f{}, v8f{}};
    float m[8], l[8];
#pragma unroll
    for (int r = 0; r < 8; ++r) { m[r] = -1e30f; l[r] = 0.0f; }

    auto issue_chunk = [&](int kb, int buf) {
        const __bf16* ks = Kb + (size_t)(kb + crow) * HD + cseg * 16;
        const __bf16* vs = Vb + (size_t)crow * NTOK + kb + cseg * 16;
        const uint32_t kd = lds_off(&Kbuf[buf][crow * KPAD + cseg * 16]);
        const uint32_t vd = lds_off(&Vbuf[buf][crow * KPAD + cseg * 16]);
        ASYNC_LD_B128(kd,                 ks);
        ASYNC_LD_B128((uint32_t)(kd + 16), ks + 8);
        ASYNC_LD_B128(vd,                 vs);
        ASYNC_LD_B128((uint32_t)(vd + 16), vs + 8);
    };

    issue_chunk(0, 0);                         // prologue: chunk 0 in flight

    const int NCHUNK = NTOK / 64;              // 16
    for (int c = 0; c < NCHUNK; ++c) {
        const int cur = c & 1;
        if (c + 1 < NCHUNK) {
            issue_chunk((c + 1) * 64, cur ^ 1);
            asm volatile("s_wait_asynccnt 0x4" ::: "memory");  // chunk c done
        } else {
            asm volatile("s_wait_asynccnt 0x0" ::: "memory");
        }
        __syncthreads();                       // chunk c visible to all waves

        const __bf16* Kl = Kbuf[cur];
        const __bf16* Vl = Vbuf[cur];

        // --- S = (Q*scale) . K^T : four 16x16 f32 tiles over 64 keys --------
        v8f S[4] = {v8f{}, v8f{}, v8f{}, v8f{}};
#pragma unroll
        for (int jt = 0; jt < 4; ++jt) {
            const __bf16* kp = Kl + (jt * 16 + l16) * KPAD + half * 16;
            v16bf b0 = ldv16_lds(kp);          // dk  0..31 rows
            v16bf b1 = ldv16_lds(kp + 32);     // dk 32..63 rows
            S[jt] = wmma_bf16(qa0.v, b0, S[jt]);
            S[jt] = wmma_bf16(qa1.v, b1, S[jt]);
        }

        // --- online softmax update (row = r + half*8, replicated x16 lanes) -
#pragma unroll
        for (int r = 0; r < 8; ++r) {
            float v = fmaxf(fmaxf(S[0][r], S[1][r]), fmaxf(S[2][r], S[3][r]));
            v = fmaxf(v, __shfl_xor(v, 1));
            v = fmaxf(v, __shfl_xor(v, 2));
            v = fmaxf(v, __shfl_xor(v, 4));
            v = fmaxf(v, __shfl_xor(v, 8));
            const float mn    = fmaxf(m[r], v);
            const float alpha = __expf(m[r] - mn);
            m[r] = mn;
#pragma unroll
            for (int t = 0; t < 4; ++t) O[t][r] *= alpha;
            float s = 0.0f;
#pragma unroll
            for (int jt = 0; jt < 4; ++jt) {
                const float p = __expf(S[jt][r] - mn);
                S[jt][r] = p;
                s += p;
            }
            s += __shfl_xor(s, 1);
            s += __shfl_xor(s, 2);
            s += __shfl_xor(s, 4);
            s += __shfl_xor(s, 8);
            l[r] = l[r] * alpha + s;
        }

        // --- transpose P (C layout) -> A layout via per-wave LDS slab -------
#pragma unroll
        for (int r = 0; r < 8; ++r) {
            const int row = r + half * 8;
#pragma unroll
            for (int jt = 0; jt < 4; ++jt)
                Pbuf[wid][row * KPAD + jt * 16 + l16] = (__bf16)S[jt][r];
        }
        asm volatile("s_wait_dscnt 0x0" ::: "memory");  // cross-lane LDS RAW
        AF pa0, pa1;    // P A-fragments for key sub-steps 0..31 / 32..63
        {
            const __bf16* pp = Pbuf[wid] + l16 * KPAD;
            pa0.h.lo = *reinterpret_cast<const v8bf*>(pp +      half * 16);
            pa0.h.hi = *reinterpret_cast<const v8bf*>(pp +  8 + half * 16);
            pa1.h.lo = *reinterpret_cast<const v8bf*>(pp + 32 + half * 16);
            pa1.h.hi = *reinterpret_cast<const v8bf*>(pp + 40 + half * 16);
        }

        // --- O += P . V  (V^T tile in LDS: contiguous B-fragments) ----------
#pragma unroll
        for (int t = 0; t < 4; ++t) {
            const __bf16* vp = Vl + (t * 16 + l16) * KPAD + half * 16;
            v16bf bv0 = ldv16_lds(vp);         // keys  0..31 of chunk
            v16bf bv1 = ldv16_lds(vp + 32);    // keys 32..63 of chunk
            O[t] = wmma_bf16(pa0.v, bv0, O[t]);
            O[t] = wmma_bf16(pa1.v, bv1, O[t]);
        }

        __syncthreads();   // all waves done with buf `cur` before its reuse
    }

    // --- normalize by softmax denominator, store bf16 A-matrix for out-proj -
#pragma unroll
    for (int r = 0; r < 8; ++r) {
        const float inv = 1.0f / l[r];
#pragma unroll
        for (int t = 0; t < 4; ++t) O[t][r] *= inv;
    }
    const int b    = bh >> 3;
    const int head = bh & 7;
#pragma unroll
    for (int t = 0; t < 4; ++t) {
        const int col = head * 64 + t * 16 + l16;
#pragma unroll
        for (int r = 0; r < 8; ++r) {
            const int token = qtile * 16 + r + half * 8;
            aout[(size_t)(b * NTOK + token) * DIM + col] = (__bf16)O[t][r];
        }
    }
}

// ---------------------------------------------------------------------------
// Output projection: [16384,512] x [512,512] + bias + residual -> fp32 out
// ---------------------------------------------------------------------------
__global__ __launch_bounds__(128)
void out_gemm_kernel(const __bf16* __restrict__ aout,
                     const __bf16* __restrict__ woutT,
                     const float*  __restrict__ bout,
                     const float*  __restrict__ x,
                     float* __restrict__ out) {
    const int lane  = threadIdx.x & 31;
    const int wid   = threadIdx.x >> 5;
    const int half  = lane >> 4;
    const int l16   = lane & 15;
    const int wg    = blockIdx.x * 4 + wid;
    const int mtile = wg >> 3;
    const int n64   = wg & 7;

    const int arow = mtile * 16 + l16;
    v8f acc[4] = {v8f{}, v8f{}, v8f{}, v8f{}};

    for (int kk = 0; kk < DIM; kk += 32) {
        AF a;
        const __bf16* ap = aout + (size_t)arow * DIM + kk;
        a.h.lo = ldv8(ap + half * 16);
        a.h.hi = ldv8(ap + 8 + half * 16);
#pragma unroll
        for (int t = 0; t < 4; ++t) {
            const int fcol = n64 * 64 + t * 16 + l16;
            v16bf b = ldv16(woutT + (size_t)fcol * DIM + kk + half * 16);
            acc[t] = wmma_bf16(a.v, b, acc[t]);
        }
    }

#pragma unroll
    for (int t = 0; t < 4; ++t) {
        const int col  = n64 * 64 + t * 16 + l16;
        const float bb = bout[col];
#pragma unroll
        for (int r = 0; r < 8; ++r) {
            const int row = mtile * 16 + r + half * 8;
            const size_t idx = (size_t)row * DIM + col;
            out[idx] = acc[t][r] + bb + x[idx];
        }
    }
}

// ---------------------------------------------------------------------------
// Launch
// ---------------------------------------------------------------------------
extern "C" void kernel_launch(void* const* d_in, const int* in_sizes, int n_in,
                              void* d_out, int out_size, void* d_ws, size_t ws_size,
                              hipStream_t stream) {
    const float* x     = (const float*)d_in[0];
    const float* w_qkv = (const float*)d_in[1];
    const float* b_qkv = (const float*)d_in[2];
    const float* w_out = (const float*)d_in[3];
    const float* b_out = (const float*)d_in[4];
    float* out = (float*)d_out;

    char* ws = (char*)d_ws;
    // bf16 staging buffers (total ~86 MB, fits the 192 MB L2)
    __bf16* xbf   = (__bf16*)(ws);                       // 16384*512*2 = 16 MB
    __bf16* wqkvT = (__bf16*)(ws + 16777216);            // 1536*512*2
    __bf16* woutT = (__bf16*)(ws + 18350080);            // 512*512*2
    __bf16* Qb    = (__bf16*)(ws + 18874368);            // [b,H,n,d]  16 MB
    __bf16* Kb    = (__bf16*)(ws + 35651584);            // [b,H,n,d]  16 MB
    __bf16* Vt    = (__bf16*)(ws + 52428800);            // [b,H,d,n]  16 MB
    __bf16* aout  = (__bf16*)(ws + 69206016);            // [b*n,512]  16 MB

    cvt_x_kernel <<<4096, 256, 0, stream>>>(x, xbf, MTOT * DIM);
    cvt_wT_kernel<<<1024, 256, 0, stream>>>(w_qkv, wqkvT, DIM, 3 * DIM);
    cvt_wT_kernel<<< 512, 256, 0, stream>>>(w_out, woutT, DIM, DIM);

    // 1024 M-tiles * 24 N64-groups = 24576 waves / 4 per block
    qkv_gemm_kernel<<<6144, 128, 0, stream>>>(xbf, wqkvT, b_qkv, Qb, Kb, Vt);
    // 128 (b,head) * 8 qtile-groups = 1024 blocks of 8 waves
    attn_kernel   <<<1024, 256, 0, stream>>>(Qb, Kb, Vt, aout);
    // 1024 M-tiles * 8 N64-groups = 8192 waves / 4 per block
    out_gemm_kernel<<<2048, 128, 0, stream>>>(aout, woutT, b_out, x, out);
}
